// AutoCompleteDecoderModel_87479893885606
// MI455X (gfx1250) — matrix operand order repaired and checked by reading, not compiled, not run on hardware
//
#include <hip/hip_runtime.h>

typedef __bf16 bf16;
typedef __attribute__((ext_vector_type(16))) __bf16 v16bf;
typedef __attribute__((ext_vector_type(8)))  __bf16 v8bf;
typedef __attribute__((ext_vector_type(8)))  float  v8f;

constexpr int kB  = 128;   // batch
constexpr int kLC = 256;   // encoder length
constexpr int kLE = 257;   // decoder length (+1)
constexpr int kH  = 512;   // hidden
constexpr int kV  = 128;   // vocab
constexpr int kG  = 4 * kH;   // 2048 (gate width)
constexpr int k2H = 2 * kH;   // 1024
constexpr int k3H = 3 * kH;   // 1536

// Build a 16-bit A fragment (16x32 tile slice for this lane) from two 16-byte loads.
__device__ __forceinline__ v16bf load_afrag(const bf16* p) {
  v8bf lo = *(const v8bf*)p;
  v8bf hi = *(const v8bf*)(p + 16);
  v16bf r;
#pragma unroll
  for (int j = 0; j < 8; ++j) { r[j] = lo[j]; r[8 + j] = hi[j]; }
  return r;
}

// ---------------------------------------------------------------------------
// bf16 WMMA GEMM:  C[M,N] = A[M,K] * Bt[N,K]^T   (Bt is "W" row-major = W.T applied)
// One wave computes a 32x64 C macro-tile: 2 A frags x 4 B frags -> 8 WMMAs/K-step.
// OUT: 0 = f32 store, 1 = f32 + bf16 store, 2 = tanh -> f32 + bf16, 3 = bf16 only
// Requires M % 32 == 0, N % 64 == 0, K % 32 == 0.
// ---------------------------------------------------------------------------
template <int OUT>
__global__ __launch_bounds__(256) void gemm_bf16_wmma(
    const bf16* __restrict__ A, int lda,
    const bf16* __restrict__ Bt,
    float* __restrict__ Cf, int ldc,
    bf16* __restrict__ Cb, int ldcb,
    int M, int N, int K)
{
  int wave  = (blockIdx.x * blockDim.x + threadIdx.x) >> 5;
  int lane  = threadIdx.x & 31;
  int tilesN = N >> 6;            // 64-wide macro-columns
  int tm = (wave / tilesN) << 5;  // 32-tall macro-rows
  int tn = (wave % tilesN) << 6;
  if (tm >= M) return;

  // A fragment (16-bit A, 16x32): lanes 0-15 rows M=0..15 with K 0-15,
  // lanes 16-31 same rows with K 16-31.
  int arow = lane & 15;
  int akb  = (lane >> 4) << 4;    // 0 or 16
  // B fragment (32x16): lanes 0-15 N=col K=0..15; lanes 16-31 N=col K=16..31.
  int bcol = lane & 15;
  int bks  = (lane >> 4) << 4;    // 0 or 16

  const bf16* Ap0 = A  + (size_t)(tm + arow) * lda;
  const bf16* Ap1 = A  + (size_t)(tm + 16 + arow) * lda;
  const bf16* Bp  = Bt + (size_t)(tn + bcol) * K;
  const size_t bstride = (size_t)16 * K;   // next 16-column B subtile

  v8f acc[8] = {};   // acc[mi*4 + ni], mi in {0,1}, ni in {0..3}
  for (int k0 = 0; k0 < K; k0 += 32) {
    v16bf a0 = *(const v16bf*)(Ap0 + k0 + akb);
    v16bf a1 = *(const v16bf*)(Ap1 + k0 + akb);
    v16bf b0 = *(const v16bf*)(Bp + 0 * bstride + k0 + bks);
    v16bf b1 = *(const v16bf*)(Bp + 1 * bstride + k0 + bks);
    v16bf b2 = *(const v16bf*)(Bp + 2 * bstride + k0 + bks);
    v16bf b3 = *(const v16bf*)(Bp + 3 * bstride + k0 + bks);
    acc[0] = __builtin_amdgcn_wmma_f32_16x16x32_bf16(false, a0, false, b0, (short)0, acc[0], false, false);
    acc[1] = __builtin_amdgcn_wmma_f32_16x16x32_bf16(false, a0, false, b1, (short)0, acc[1], false, false);
    acc[2] = __builtin_amdgcn_wmma_f32_16x16x32_bf16(false, a0, false, b2, (short)0, acc[2], false, false);
    acc[3] = __builtin_amdgcn_wmma_f32_16x16x32_bf16(false, a0, false, b3, (short)0, acc[3], false, false);
    acc[4] = __builtin_amdgcn_wmma_f32_16x16x32_bf16(false, a1, false, b0, (short)0, acc[4], false, false);
    acc[5] = __builtin_amdgcn_wmma_f32_16x16x32_bf16(false, a1, false, b1, (short)0, acc[5], false, false);
    acc[6] = __builtin_amdgcn_wmma_f32_16x16x32_bf16(false, a1, false, b2, (short)0, acc[6], false, false);
    acc[7] = __builtin_amdgcn_wmma_f32_16x16x32_bf16(false, a1, false, b3, (short)0, acc[7], false, false);
  }

  // C/D layout: lanes 0-15 N=lane, M=vgpr; lanes 16-31 N=lane-16, M=8+vgpr.
  int cn  = lane & 15;
  int cmb = (lane >> 4) << 3;
#pragma unroll
  for (int mi = 0; mi < 2; ++mi) {
#pragma unroll
    for (int ni = 0; ni < 4; ++ni) {
      const v8f& a = acc[mi * 4 + ni];
#pragma unroll
      for (int r = 0; r < 8; ++r) {
        float v = a[r];
        if (OUT == 2) v = tanhf(v);
        int m  = tm + mi * 16 + cmb + r;
        int nn = tn + ni * 16 + cn;
        if (OUT != 3) Cf[(size_t)m * ldc + nn] = v;
        if (OUT != 0) Cb[(size_t)m * ldcb + nn] = (bf16)v;
      }
    }
  }
}

// ---------------------------------------------------------------------------
// LSTM gate nonlinearity + state update.
// G[b, 4H] = h_prev @ Whh.T (from WMMA GEMM);  x one-hot -> WihT[tok] row gather.
// ---------------------------------------------------------------------------
__global__ __launch_bounds__(256) void lstm_update(
    const float* __restrict__ G, const float* __restrict__ WihT,
    const float* __restrict__ bias, const int* __restrict__ idx,
    int idx_stride, int pos,
    float* __restrict__ c, float* __restrict__ h_f32,
    bf16* __restrict__ h_bf, int h_bf_stride,
    bf16* __restrict__ enc_out, int enc_pos, int enc_off)
{
  int tid = blockIdx.x * blockDim.x + threadIdx.x;   // B*H threads
  int b = tid >> 9;
  int n = tid & (kH - 1);
  int tok = idx[b * idx_stride + pos];
  const float* g  = G + (size_t)b * kG;
  const float* wx = WihT + (size_t)tok * kG;
  float gi = g[n]            + wx[n]            + bias[n];
  float gf = g[kH + n]       + wx[kH + n]       + bias[kH + n];
  float gg = g[2 * kH + n]   + wx[2 * kH + n]   + bias[2 * kH + n];
  float go = g[3 * kH + n]   + wx[3 * kH + n]   + bias[3 * kH + n];
  float si = 1.f / (1.f + expf(-gi));
  float sf = 1.f / (1.f + expf(-gf));
  float so = 1.f / (1.f + expf(-go));
  float cn = sf * c[tid] + si * tanhf(gg);
  float hn = so * tanhf(cn);
  c[tid] = cn;
  h_f32[tid] = hn;
  h_bf[(size_t)b * h_bf_stride + n] = (bf16)hn;
  if (enc_out)
    enc_out[((size_t)b * kLC + enc_pos) * k2H + enc_off + n] = (bf16)hn;
}

// ---------------------------------------------------------------------------
// Attention: scores = enc_proj[b] @ h[b], softmax (masked), ctx = attn @ enc[b].
// Also builds A_out = [h | ctx] in bf16 for the Wout GEMM.
// One block per batch row, 256 threads (= LC).
// ---------------------------------------------------------------------------
__global__ __launch_bounds__(256) void attention_ctx(
    const bf16* __restrict__ encproj, const bf16* __restrict__ enc,
    const float* __restrict__ h, const unsigned char* __restrict__ C_pad,
    float* __restrict__ attn, float* __restrict__ ctx, bf16* __restrict__ A_out)
{
  int b = blockIdx.x, tid = threadIdx.x;
  __shared__ float s_attn[kLC];
  __shared__ float red[kLC];

  const bf16*  ep = encproj + ((size_t)b * kLC + tid) * kH;
  const float* hb = h + (size_t)b * kH;
  float s = 0.f;
  for (int k = 0; k < kH; ++k) s += (float)ep[k] * hb[k];
  if (C_pad[b * kLC + tid]) s = -3.0e38f;

  red[tid] = s; __syncthreads();
  for (int o = 128; o > 0; o >>= 1) { if (tid < o) red[tid] = fmaxf(red[tid], red[tid + o]); __syncthreads(); }
  float m = red[0]; __syncthreads();
  float e = expf(s - m);
  red[tid] = e; __syncthreads();
  for (int o = 128; o > 0; o >>= 1) { if (tid < o) red[tid] += red[tid + o]; __syncthreads(); }
  float a = e / red[0];
  s_attn[tid] = a;
  attn[b * kLC + tid] = a;
  __syncthreads();

  const bf16* eb = enc + (size_t)b * kLC * k2H;
  for (int col = tid; col < k2H; col += 256) {
    float acc = 0.f;
    for (int l = 0; l < kLC; ++l) acc += s_attn[l] * (float)eb[(size_t)l * k2H + col];
    ctx[(size_t)b * k2H + col] = acc;
    A_out[(size_t)b * k3H + kH + col] = (bf16)acc;
  }
  for (int j = tid; j < kH; j += 256)
    A_out[(size_t)b * k3H + j] = (bf16)hb[j];
}

// ---------------------------------------------------------------------------
// Final per-step: vocab softmax, p_gen, pointer-copy mix, NLL for target token.
// One block per batch row, 256 threads.
// ---------------------------------------------------------------------------
__global__ __launch_bounds__(256) void final_step(
    const float* __restrict__ logits, const float* __restrict__ ctx,
    const float* __restrict__ h, const float* __restrict__ c,
    const float* __restrict__ prev_out, const float* __restrict__ attn,
    const int* __restrict__ C_idx, const int* __restrict__ E_idx, int t,
    const float* __restrict__ pg_ctx, const float* __restrict__ pg_in,
    const float* __restrict__ pg_h, const float* __restrict__ pg_c,
    const float* __restrict__ pg_b, float* __restrict__ nll)
{
  int b = blockIdx.x, tid = threadIdx.x;
  __shared__ float red[256];
  const float* lg = logits + (size_t)b * kV;

  float lv = (tid < kV) ? lg[tid] : -3.0e38f;
  red[tid] = lv; __syncthreads();
  for (int o = 128; o > 0; o >>= 1) { if (tid < o) red[tid] = fmaxf(red[tid], red[tid + o]); __syncthreads(); }
  float m = red[0]; __syncthreads();
  float e = (tid < kV) ? expf(lv - m) : 0.f;
  red[tid] = e; __syncthreads();
  for (int o = 128; o > 0; o >>= 1) { if (tid < o) red[tid] += red[tid + o]; __syncthreads(); }
  float denom = red[0]; __syncthreads();

  // p_gen = sigmoid(ctx.pg_ctx + nin.pg_in + h.pg_h + c.pg_c + pg_b)
  float p = 0.f;
  const float* cx  = ctx + (size_t)b * k2H;
  const float* hb_ = h + (size_t)b * kH;
  const float* cb_ = c + (size_t)b * kH;
  const float* po  = prev_out + (size_t)b * kH;
  for (int j = tid; j < k2H; j += 256) p += cx[j] * pg_ctx[j];
  for (int j = tid; j < kH; j += 256)
    p += po[j] * pg_in[kV + j] + hb_[j] * pg_h[j] + cb_[j] * pg_c[j];
  red[tid] = p; __syncthreads();
  for (int o = 128; o > 0; o >>= 1) { if (tid < o) red[tid] += red[tid + o]; __syncthreads(); }
  int xtok = E_idx[b * kLE + t];
  int tgt  = E_idx[b * kLE + t + 1];
  float z = red[0] + pg_in[xtok] + pg_b[0];
  float pgen = 1.f / (1.f + expf(-z));
  __syncthreads();

  // pointer-copy mass at target token
  float cp = (C_idx[b * kLC + tid] == tgt) ? attn[b * kLC + tid] : 0.f;
  red[tid] = cp; __syncthreads();
  for (int o = 128; o > 0; o >>= 1) { if (tid < o) red[tid] += red[tid + o]; __syncthreads(); }

  if (tid == 0) {
    float gt = expf(lg[tgt] - m) / denom;
    float prob = gt * pgen + (1.f - pgen) * red[0];
    nll[(size_t)b * (kLE - 1) + t] = (tgt == 0) ? 0.f : -logf(prob);
  }
}

// ----------------------------- small utilities -----------------------------
__global__ __launch_bounds__(256) void f32_to_bf16_k(const float* __restrict__ in,
                                                     bf16* __restrict__ out, int n)
{
  int i = blockIdx.x * blockDim.x + threadIdx.x;
  if (i < n) out[i] = (bf16)in[i];
}

__global__ __launch_bounds__(256) void transpose_cols_k(const float* __restrict__ in,
                                                        int in_stride, int R, int C,
                                                        float* __restrict__ out)
{
  int i = blockIdx.x * blockDim.x + threadIdx.x;
  if (i >= R * C) return;
  int r = i / C, cc = i - r * C;
  out[(size_t)cc * R + r] = in[(size_t)r * in_stride + cc];
}

// Bdec[n, 0:512] = dec_Wih[n, V:V+H];  Bdec[n, 512:1024] = dec_Whh[n, :]
__global__ __launch_bounds__(256) void build_bdec_k(const float* __restrict__ dec_Wih,
                                                    const float* __restrict__ dec_Whh,
                                                    bf16* __restrict__ Bdec)
{
  int i = blockIdx.x * blockDim.x + threadIdx.x;   // kG * k2H threads
  int n = i >> 10, k = i & (k2H - 1);
  float v = (k < kH) ? dec_Wih[(size_t)n * (kH + kV) + kV + k]
                     : dec_Whh[(size_t)n * kH + (k - kH)];
  Bdec[i] = (bf16)v;
}

__global__ __launch_bounds__(256) void pack_cat_k(const bf16* __restrict__ hf,
                                                  const bf16* __restrict__ hb,
                                                  const float* __restrict__ cf,
                                                  const float* __restrict__ cb,
                                                  bf16* __restrict__ hcat,
                                                  bf16* __restrict__ ccat)
{
  int i = blockIdx.x * blockDim.x + threadIdx.x;   // B*H threads
  int b = i >> 9, n = i & (kH - 1);
  hcat[(size_t)b * k2H + n]      = hf[i];
  hcat[(size_t)b * k2H + kH + n] = hb[i];
  ccat[(size_t)b * k2H + n]      = (bf16)cf[i];
  ccat[(size_t)b * k2H + kH + n] = (bf16)cb[i];
}

__global__ __launch_bounds__(256) void zero_u32_k(unsigned int* __restrict__ p, int n)
{
  int i = blockIdx.x * blockDim.x + threadIdx.x;
  if (i < n) p[i] = 0u;
}

// ---------------------------------------------------------------------------
extern "C" void kernel_launch(void* const* d_in, const int* in_sizes, int n_in,
                              void* d_out, int out_size, void* d_ws, size_t ws_size,
                              hipStream_t stream)
{
  (void)in_sizes; (void)n_in; (void)out_size; (void)ws_size;

  const int*   C_idx = (const int*)d_in[0];
  const int*   E_idx = (const int*)d_in[1];
  const unsigned char* C_pad = (const unsigned char*)d_in[2];
  const float* enc_Wih_f = (const float*)d_in[3];
  const float* enc_Whh_f = (const float*)d_in[4];
  const float* enc_b_f   = (const float*)d_in[5];
  const float* enc_Wih_b = (const float*)d_in[6];
  const float* enc_Whh_b = (const float*)d_in[7];
  const float* enc_b_b   = (const float*)d_in[8];
  const float* dec_Wih   = (const float*)d_in[9];
  const float* dec_Whh   = (const float*)d_in[10];
  const float* dec_b     = (const float*)d_in[11];
  const float* Wh        = (const float*)d_in[12];
  const float* Wc        = (const float*)d_in[13];
  const float* Wattn     = (const float*)d_in[14];
  const float* Wout      = (const float*)d_in[15];
  const float* Wvocab    = (const float*)d_in[16];
  const float* pg_ctx    = (const float*)d_in[17];
  const float* pg_in     = (const float*)d_in[18];
  const float* pg_h      = (const float*)d_in[19];
  const float* pg_c      = (const float*)d_in[20];
  const float* pg_b      = (const float*)d_in[21];
  float* nll = (float*)d_out;

  char* base = (char*)d_ws;
  size_t off = 0;
  auto alloc = [&](size_t bytes) -> void* {
    void* p = base + off;
    off = (off + bytes + 255) & ~(size_t)255;
    return p;
  };

  // ---- workspace layout (bf16 activations/weights, f32 states) ----
  bf16* enc_bf     = (bf16*)alloc((size_t)kB * kLC * k2H * 2);   // 64 MB
  bf16* encproj_bf = (bf16*)alloc((size_t)kB * kLC * kH * 2);    // 32 MB
  bf16* Whh_f_bf   = (bf16*)alloc((size_t)kG * kH * 2);
  bf16* Whh_b_bf   = (bf16*)alloc((size_t)kG * kH * 2);
  bf16* Bdec_bf    = (bf16*)alloc((size_t)kG * k2H * 2);
  bf16* Wattn_bf   = (bf16*)alloc((size_t)kH * k2H * 2);
  bf16* Wout_bf    = (bf16*)alloc((size_t)kH * k3H * 2);
  bf16* Wvocab_bf  = (bf16*)alloc((size_t)kV * kH * 2);
  bf16* Wh_bf      = (bf16*)alloc((size_t)kH * k2H * 2);
  bf16* Wc_bf      = (bf16*)alloc((size_t)kH * k2H * 2);
  float* WihT_f    = (float*)alloc((size_t)kV * kG * 4);
  float* WihT_b    = (float*)alloc((size_t)kV * kG * 4);
  float* WihT_dec  = (float*)alloc((size_t)kV * kG * 4);
  float* Gf        = (float*)alloc((size_t)kB * kG * 4);
  float* Gb        = (float*)alloc((size_t)kB * kG * 4);
  float* Gdec      = (float*)alloc((size_t)kB * kG * 4);
  bf16* hf_bf      = (bf16*)alloc((size_t)kB * kH * 2);
  bf16* hb_bf      = (bf16*)alloc((size_t)kB * kH * 2);
  float* cf        = (float*)alloc((size_t)kB * kH * 4);
  float* cb        = (float*)alloc((size_t)kB * kH * 4);
  float* h_scr     = (float*)alloc((size_t)kB * kH * 4);
  bf16* hcat_bf    = (bf16*)alloc((size_t)kB * k2H * 2);
  bf16* ccat_bf    = (bf16*)alloc((size_t)kB * k2H * 2);
  float* h_dec     = (float*)alloc((size_t)kB * kH * 4);
  float* c_dec     = (float*)alloc((size_t)kB * kH * 4);
  bf16* A_cat      = (bf16*)alloc((size_t)kB * k2H * 2);   // [prev_out | h]
  float* tout0     = (float*)alloc((size_t)kB * kH * 4);
  float* tout1     = (float*)alloc((size_t)kB * kH * 4);
  float* attn      = (float*)alloc((size_t)kB * kLC * 4);
  float* ctx       = (float*)alloc((size_t)kB * k2H * 4);
  bf16* A_out      = (bf16*)alloc((size_t)kB * k3H * 2);   // [h | ctx]
  float* logits    = (float*)alloc((size_t)kB * kV * 4);

  auto cvt = [&](const float* in, bf16* out, int n) {
    f32_to_bf16_k<<<(n + 255) / 256, 256, 0, stream>>>(in, out, n);
  };
  auto zero = [&](void* p, size_t bytes) {
    int words = (int)(bytes / 4);
    zero_u32_k<<<(words + 255) / 256, 256, 0, stream>>>((unsigned int*)p, words);
  };
  auto gemm = [&](int mode, const bf16* A, int lda, const bf16* Bt,
                  float* Cf, int ldc, bf16* Cb, int ldcb, int M, int N, int K) {
    int waves = (M / 32) * (N / 64);
    int blocks = (waves + 7) / 8;
    switch (mode) {
      case 0: gemm_bf16_wmma<0><<<blocks, 256, 0, stream>>>(A, lda, Bt, Cf, ldc, Cb, ldcb, M, N, K); break;
      case 1: gemm_bf16_wmma<1><<<blocks, 256, 0, stream>>>(A, lda, Bt, Cf, ldc, Cb, ldcb, M, N, K); break;
      case 2: gemm_bf16_wmma<2><<<blocks, 256, 0, stream>>>(A, lda, Bt, Cf, ldc, Cb, ldcb, M, N, K); break;
      default: gemm_bf16_wmma<3><<<blocks, 256, 0, stream>>>(A, lda, Bt, Cf, ldc, Cb, ldcb, M, N, K); break;
    }
  };

  // ---- one-time weight preparation ----
  cvt(enc_Whh_f, Whh_f_bf, kG * kH);
  cvt(enc_Whh_b, Whh_b_bf, kG * kH);
  cvt(Wattn,     Wattn_bf, kH * k2H);
  cvt(Wout,      Wout_bf,  kH * k3H);
  cvt(Wvocab,    Wvocab_bf, kV * kH);
  cvt(Wh,        Wh_bf,    kH * k2H);
  cvt(Wc,        Wc_bf,    kH * k2H);
  build_bdec_k<<<(kG * k2H + 255) / 256, 256, 0, stream>>>(dec_Wih, dec_Whh, Bdec_bf);
  transpose_cols_k<<<(kG * kV + 255) / 256, 256, 0, stream>>>(enc_Wih_f, kV, kG, kV, WihT_f);
  transpose_cols_k<<<(kG * kV + 255) / 256, 256, 0, stream>>>(enc_Wih_b, kV, kG, kV, WihT_b);
  transpose_cols_k<<<(kG * kV + 255) / 256, 256, 0, stream>>>(dec_Wih, kH + kV, kG, kV, WihT_dec);

  // ---- zero initial states (poison-safe: re-init every call) ----
  zero(hf_bf, (size_t)kB * kH * 2);
  zero(hb_bf, (size_t)kB * kH * 2);
  zero(cf, (size_t)kB * kH * 4);
  zero(cb, (size_t)kB * kH * 4);
  zero(A_cat, (size_t)kB * k2H * 2);
  zero(tout0, (size_t)kB * kH * 4);
  zero(tout1, (size_t)kB * kH * 4);

  // ---- bidirectional encoder: 256 sequential steps ----
  for (int t = 0; t < kLC; ++t) {
    gemm(0, hf_bf, kH, Whh_f_bf, Gf, kG, nullptr, 0, kB, kG, kH);
    gemm(0, hb_bf, kH, Whh_b_bf, Gb, kG, nullptr, 0, kB, kG, kH);
    lstm_update<<<kB * kH / 256, 256, 0, stream>>>(
        Gf, WihT_f, enc_b_f, C_idx, kLC, t,
        cf, h_scr, hf_bf, kH, enc_bf, t, 0);
    lstm_update<<<kB * kH / 256, 256, 0, stream>>>(
        Gb, WihT_b, enc_b_b, C_idx, kLC, kLC - 1 - t,
        cb, h_scr, hb_bf, kH, enc_bf, kLC - 1 - t, kH);
  }

  // ---- decoder init: h0 = [hf,hb]@Wh.T ; c0 = [cf,cb]@Wc.T ; enc_proj ----
  pack_cat_k<<<kB * kH / 256, 256, 0, stream>>>(hf_bf, hb_bf, cf, cb, hcat_bf, ccat_bf);
  gemm(1, hcat_bf, k2H, Wh_bf, h_dec, kH, A_cat + kH, k2H, kB, kH, k2H);
  gemm(0, ccat_bf, k2H, Wc_bf, c_dec, kH, nullptr, 0, kB, kH, k2H);
  gemm(3, enc_bf, k2H, Wattn_bf, nullptr, 0, encproj_bf, kH, kB * kLC, kH, k2H);

  // ---- decoder: 256 sequential attention steps ----
  for (int t = 0; t < kLE - 1; ++t) {
    float* tout_cur  = (t & 1) ? tout1 : tout0;
    float* tout_prev = (t & 1) ? tout0 : tout1;
    // gates = [prev_out | h] @ [dec_Wih[:,V:] | dec_Whh]^T
    gemm(0, A_cat, k2H, Bdec_bf, Gdec, kG, nullptr, 0, kB, kG, k2H);
    lstm_update<<<kB * kH / 256, 256, 0, stream>>>(
        Gdec, WihT_dec, dec_b, E_idx, kLE, t,
        c_dec, h_dec, A_cat + kH, k2H, nullptr, 0, 0);
    attention_ctx<<<kB, 256, 0, stream>>>(encproj_bf, enc_bf, h_dec, C_pad,
                                          attn, ctx, A_out);
    // t_out = tanh([h|ctx] @ Wout.T); bf16 copy becomes next prev_out in A_cat
    gemm(2, A_out, k3H, Wout_bf, tout_cur, kH, A_cat, k2H, kB, kH, k3H);
    // vocab logits from the bf16 t_out just written into A_cat[:, 0:H]
    gemm(0, A_cat, k2H, Wvocab_bf, logits, kV, nullptr, 0, kB, kV, kH);
    final_step<<<kB, 256, 0, stream>>>(logits, ctx, h_dec, c_dec, tout_prev, attn,
                                       C_idx, E_idx, t,
                                       pg_ctx, pg_in, pg_h, pg_c, pg_b, nll);
  }
}
